// Attention_11046655885519
// MI455X (gfx1250) — matrix operand order repaired and checked by reading, not compiled, not run on hardware
//
#include <hip/hip_runtime.h>
#include <hip/hip_bf16.h>

// Problem constants (from reference)
#define BATCH   4
#define CDIM    128
#define HEADS_  8
#define HD_     16        // v depth per head
#define KD_     8         // q/k depth per head
#define HCH     256       // qkv channels = 128 + 2*8*8
#define IMGH    40
#define IMGW    40
#define NTOK    1600      // 40*40
#define EPS_    1e-3f
#define SCALE_  0.35355339059327373f  // 8^-0.5

typedef __attribute__((ext_vector_type(16))) _Float16 v16h;
typedef __attribute__((ext_vector_type(8)))  float    v8f;

// ---------------------------------------------------------------------------
// Kernel 1: qkv = BN(conv1x1(x)) as a WMMA GEMM.
// M=256 out-ch, K=128 in-ch (4 chunks of 32), N=1600 tokens per batch.
// One wave per 16x16 output tile; 6400 tiles -> 800 blocks x 8 waves.
// A-operand = weight tile (f16 on the fly), B-operand = x tile (f16 on the fly).
// ---------------------------------------------------------------------------
__global__ __launch_bounds__(256)
void qkv_wmma_kernel(const float* __restrict__ x, const float* __restrict__ w,
                     const float* __restrict__ g, const float* __restrict__ bb,
                     const float* __restrict__ mm, const float* __restrict__ vv,
                     float* __restrict__ q32, _Float16* __restrict__ q16) {
  const int wave  = threadIdx.x >> 5;
  const int lane  = threadIdx.x & 31;
  const int laneM = lane & 15;
  const int grp   = lane >> 4;
  const int tile  = blockIdx.x * 8 + wave;       // [0, 4*16*100)
  const int b     = tile / 1600;
  const int rem   = tile % 1600;
  const int m0    = (rem / 100) * 16;            // out-channel tile
  const int n0    = (rem % 100) * 16;            // token tile
  const float* xb = x + (size_t)b * CDIM * NTOK;

  v8f acc = {};
#pragma unroll
  for (int kc = 0; kc < 4; ++kc) {               // K = 128 = 4 x 32
    v16h af, bf;
#pragma unroll
    for (int idx = 0; idx < 16; ++idx) {
      int r = idx >> 1, hh = idx & 1;
      int kA = (r & 3) * 2 + hh + (r >> 2) * 16 + grp * 8 + kc * 32;
      af[idx] = (_Float16)w[(m0 + laneM) * CDIM + kA];
      int kB = (r & 3) * 2 + hh + (r >> 2) * 16 + grp * 8 + kc * 32;
      bf[idx] = (_Float16)xb[(size_t)kB * NTOK + n0 + laneM];
    }
    acc = __builtin_amdgcn_wmma_f32_16x16x32_f16(
        false, af, false, bf, (short)0, acc, false, false);
  }
#pragma unroll
  for (int r = 0; r < 8; ++r) {
    int c = m0 + r + 8 * grp;
    float s = g[c] * rsqrtf(vv[c] + EPS_);
    float y = (acc[r] - mm[c]) * s + bb[c];
    size_t o = ((size_t)b * HCH + c) * NTOK + n0 + laneM;
    q32[o] = y;
    q16[o] = (_Float16)y;
  }
}

// ---------------------------------------------------------------------------
// Kernel 2: fused softmax attention per (b,h) using v_wmma_f32_16x16x32_f16.
// grid = (4 splits, 8 heads, 4 batch), block = 256 (8 waves).
// V is staged into LDS with GLOBAL_LOAD_ASYNC_TO_LDS_B128 (ASYNCcnt path).
// ---------------------------------------------------------------------------
__global__ __launch_bounds__(256)
void attn_kernel(const _Float16* __restrict__ qkv16, float* __restrict__ aout) {
  const int split = blockIdx.x;       // 0..3
  const int h     = blockIdx.y;       // 0..7
  const int b     = blockIdx.z;       // 0..3
  const int tid   = threadIdx.x;
  const int wave  = tid >> 5;
  const int lane  = tid & 31;
  const int laneM = lane & 15;        // N index of frags
  const int grp   = lane >> 4;        // lane half-group

  __shared__ _Float16 Vsh[HD_ * NTOK];     // 51200 B  [d][j]
  __shared__ _Float16 Pbuf[8][16 * 32];    //  8192 B  per-wave [i][j_local]
  __shared__ float    RSsh[8][16];         //   512 B  per-wave row sums

  const _Float16* base = qkv16 + (size_t)((b * HEADS_ + h) * 32) * NTOK;
  const _Float16* Qg   = base;                 // channels 0..7
  const _Float16* Kg   = base + 8 * NTOK;      // channels 8..15

  // ---- async copy V (channels 16..31, contiguous 51200 B) into LDS --------
  {
    const _Float16* vsrc = base + 16 * NTOK;
    unsigned lds0 = (unsigned)(size_t)(void*)&Vsh[0];
    for (int idx = tid; idx < (HD_ * NTOK * 2) / 16; idx += 256) {  // 3200 x 16B
      unsigned lds_addr = lds0 + idx * 16;
      unsigned voff     = (unsigned)(idx * 16);
      asm volatile("global_load_async_to_lds_b128 %0, %1, %2"
                   :: "v"(lds_addr), "v"(voff), "s"(vsrc)
                   : "memory");
    }
    asm volatile("s_wait_asynccnt 0x0" ::: "memory");
  }
  __syncthreads();

  for (int qt = wave + 8 * split; qt < 100; qt += 32) {
    const int i0 = qt * 16;
    __builtin_prefetch(Kg + (lane & 7) * NTOK + i0, 0, 1);  // warm K rows (L2)

    // A-operand: Q^T tile, 16 tokens x 32 (KD=8 real, rest zero-padded)
    v16h aq;
#pragma unroll
    for (int idx = 0; idx < 16; ++idx) {
      int r = idx >> 1, hh = idx & 1;
      int kd = (r & 3) * 2 + hh + (r >> 2) * 16 + grp * 8;
      aq[idx] = (kd < KD_) ? Qg[kd * NTOK + i0 + laneM] : (_Float16)0.0f;
    }

    v8f acc = {};                 // output frag: M=d, N=query
    float rs[8];
#pragma unroll
    for (int r = 0; r < 8; ++r) rs[r] = 0.0f;

    for (int jc = 0; jc < 50; ++jc) {          // 32 keys per chunk
#pragma unroll
      for (int jhalf = 0; jhalf < 2; ++jhalf) {
        const int j0 = jc * 32 + jhalf * 16;
        // B-operand: K tile (KD=8 rows, zero-padded to 32)
        v16h bk;
#pragma unroll
        for (int idx = 0; idx < 16; ++idx) {
          int r = idx >> 1, hh = idx & 1;
          int kd = (r & 3) * 2 + hh + (r >> 2) * 16 + grp * 8;
          bk[idx] = (kd < KD_) ? Kg[kd * NTOK + j0 + laneM] : (_Float16)0.0f;
        }
        v8f zero = {};
        v8f s = __builtin_amdgcn_wmma_f32_16x16x32_f16(
            false, aq, false, bk, (short)0, zero, false, false);
        // exp + row-sum (softmax w/o max-sub: |S| is small by construction)
#pragma unroll
        for (int r = 0; r < 8; ++r) {
          float p = __expf(s[r] * SCALE_);
          float rowp = p;
#pragma unroll
          for (int m = 1; m < 16; m <<= 1) rowp += __shfl_xor(rowp, m, 16);
          rs[r] += rowp;
          Pbuf[wave][(r + 8 * grp) * 32 + jhalf * 16 + laneM] = (_Float16)p;
        }
      }
      asm volatile("s_wait_dscnt 0" ::: "memory");  // per-wave staging visible
      // A-operand: V (16 x 32 from LDS), B-operand: P^T (read from staging)
      v16h av, bp;
#pragma unroll
      for (int idx = 0; idx < 16; ++idx) {
        int r = idx >> 1, hh = idx & 1;
        int jA = (r & 3) * 2 + hh + (r >> 2) * 16 + grp * 8;
        av[idx] = Vsh[laneM * NTOK + jc * 32 + jA];
        int jB = (r & 3) * 2 + hh + (r >> 2) * 16 + grp * 8;
        bp[idx] = Pbuf[wave][laneM * 32 + jB];
      }
      acc = __builtin_amdgcn_wmma_f32_16x16x32_f16(
          false, av, false, bp, (short)0, acc, false, false);
    }

    // broadcast per-query row sums, normalize, store
    if (laneM == 0) {
#pragma unroll
      for (int r = 0; r < 8; ++r) RSsh[wave][r + 8 * grp] = rs[r];
    }
    asm volatile("s_wait_dscnt 0" ::: "memory");
    float inv = 1.0f / RSsh[wave][laneM];
#pragma unroll
    for (int r = 0; r < 8; ++r) {
      int d = r + 8 * grp;
      aout[(size_t)(b * CDIM + h * HD_ + d) * NTOK + i0 + laneM] = acc[r] * inv;
    }
  }
}

// ---------------------------------------------------------------------------
// Kernel 3: t = attn_out + BN(dwconv3x3(v)); stencil stays on f32 VALU,
// emits f16 for the proj WMMA input.
// ---------------------------------------------------------------------------
__global__ __launch_bounds__(256)
void pe_add_kernel(const float* __restrict__ qkv32, const float* __restrict__ aout,
                   const float* __restrict__ pw, const float* __restrict__ g,
                   const float* __restrict__ bb, const float* __restrict__ mm,
                   const float* __restrict__ vv, _Float16* __restrict__ t16) {
  int t = blockIdx.x * 256 + threadIdx.x;          // [0, 4*128*1600)
  int n = t % NTOK;
  int c = (t / NTOK) % CDIM;
  int b = t / (NTOK * CDIM);
  int yy = n / IMGW, xx = n % IMGW;
  int hh = c / HD_, d = c % HD_;
  const float* vplane = qkv32 + (size_t)((b * HEADS_ + hh) * 32 + 16 + d) * NTOK;
  float acc = 0.f;
#pragma unroll
  for (int ky = -1; ky <= 1; ++ky)
#pragma unroll
    for (int kx = -1; kx <= 1; ++kx) {
      int y2 = yy + ky, x2 = xx + kx;
      if (y2 >= 0 && y2 < IMGH && x2 >= 0 && x2 < IMGW)
        acc += pw[c * 9 + (ky + 1) * 3 + (kx + 1)] * vplane[y2 * IMGW + x2];
    }
  float s = g[c] * rsqrtf(vv[c] + EPS_);
  t16[t] = (_Float16)(aout[t] + (acc - mm[c]) * s + bb[c]);
}

// ---------------------------------------------------------------------------
// Kernel 4: y = BN(conv1x1(t, proj_w)) as a WMMA GEMM.
// M=128, K=128 (4 chunks of 32), N=1600 per batch. 3200 tiles -> 400 blocks.
// ---------------------------------------------------------------------------
__global__ __launch_bounds__(256)
void proj_wmma_kernel(const _Float16* __restrict__ t16, const float* __restrict__ w,
                      const float* __restrict__ g, const float* __restrict__ bb,
                      const float* __restrict__ mm, const float* __restrict__ vv,
                      float* __restrict__ out) {
  const int wave  = threadIdx.x >> 5;
  const int lane  = threadIdx.x & 31;
  const int laneM = lane & 15;
  const int grp   = lane >> 4;
  const int tile  = blockIdx.x * 8 + wave;       // [0, 4*8*100)
  const int b     = tile / 800;
  const int rem   = tile % 800;
  const int m0    = (rem / 100) * 16;
  const int n0    = (rem % 100) * 16;
  const _Float16* tb = t16 + (size_t)b * CDIM * NTOK;

  v8f acc = {};
#pragma unroll
  for (int kc = 0; kc < 4; ++kc) {
    v16h af, bf;
#pragma unroll
    for (int idx = 0; idx < 16; ++idx) {
      int r = idx >> 1, hh = idx & 1;
      int kA = (r & 3) * 2 + hh + (r >> 2) * 16 + grp * 8 + kc * 32;
      af[idx] = (_Float16)w[(m0 + laneM) * CDIM + kA];
      int kB = (r & 3) * 2 + hh + (r >> 2) * 16 + grp * 8 + kc * 32;
      bf[idx] = tb[(size_t)kB * NTOK + n0 + laneM];
    }
    acc = __builtin_amdgcn_wmma_f32_16x16x32_f16(
        false, af, false, bf, (short)0, acc, false, false);
  }
#pragma unroll
  for (int r = 0; r < 8; ++r) {
    int c = m0 + r + 8 * grp;
    float s = g[c] * rsqrtf(vv[c] + EPS_);
    out[((size_t)b * CDIM + c) * NTOK + n0 + laneM] = (acc[r] - mm[c]) * s + bb[c];
  }
}

// ---------------------------------------------------------------------------
extern "C" void kernel_launch(void* const* d_in, const int* in_sizes, int n_in,
                              void* d_out, int out_size, void* d_ws, size_t ws_size,
                              hipStream_t stream) {
  const float* x      = (const float*)d_in[0];
  const float* qkv_w  = (const float*)d_in[1];
  const float* qkv_g  = (const float*)d_in[2];
  const float* qkv_b  = (const float*)d_in[3];
  const float* qkv_m  = (const float*)d_in[4];
  const float* qkv_v  = (const float*)d_in[5];
  const float* pe_w   = (const float*)d_in[6];
  const float* pe_g   = (const float*)d_in[7];
  const float* pe_b   = (const float*)d_in[8];
  const float* pe_m   = (const float*)d_in[9];
  const float* pe_v   = (const float*)d_in[10];
  const float* proj_w = (const float*)d_in[11];
  const float* proj_g = (const float*)d_in[12];
  const float* proj_b = (const float*)d_in[13];
  const float* proj_m = (const float*)d_in[14];
  const float* proj_v = (const float*)d_in[15];

  // Workspace layout (~14.7 MB total)
  const size_t QKV_ELEMS = (size_t)BATCH * HCH * NTOK;   // 1,638,400
  const size_t OUT_ELEMS = (size_t)BATCH * CDIM * NTOK;  //   819,200
  char* ws = (char*)d_ws;
  float*    qkv32 = (float*)ws;                                   ws += QKV_ELEMS * 4;
  _Float16* qkv16 = (_Float16*)ws;                                ws += QKV_ELEMS * 2;
  float*    aout  = (float*)ws;                                   ws += OUT_ELEMS * 4;
  _Float16* t16   = (_Float16*)ws;

  qkv_wmma_kernel<<<800, 256, 0, stream>>>(
      x, qkv_w, qkv_g, qkv_b, qkv_m, qkv_v, qkv32, qkv16);

  attn_kernel<<<dim3(4, HEADS_, BATCH), 256, 0, stream>>>(qkv16, aout);

  pe_add_kernel<<<(int)(OUT_ELEMS / 256), 256, 0, stream>>>(
      qkv32, aout, pe_w, pe_g, pe_b, pe_m, pe_v, t16);

  proj_wmma_kernel<<<400, 256, 0, stream>>>(
      t16, proj_w, proj_g, proj_b, proj_m, proj_v, (float*)d_out);
}
